// TextMaskedMultiheadSelfAttention_88278757802431
// MI455X (gfx1250) — matrix operand order repaired and checked by reading, not compiled, not run on hardware
//
#include <hip/hip_runtime.h>

// ---- CDNA5 (gfx1250) wave32 WMMA types ----
typedef __attribute__((ext_vector_type(16))) __bf16 v16bf;
typedef __attribute__((ext_vector_type(8)))  __bf16 v8bf;
typedef __attribute__((ext_vector_type(8)))  float  v8f;

#define Bz 2
#define Sz 2048
#define Ez 512
#define Hz 8
#define Dz 64
#define ROWS (Bz*Sz)   // 4096

__device__ __forceinline__ __bf16 f2bf(float f) {
    union { float f; unsigned u; } a; a.f = f;
    unsigned r = (a.u + 0x7FFFu + ((a.u >> 16) & 1u)) >> 16;
    union { unsigned short s; __bf16 b; } o; o.s = (unsigned short)r;
    return o.b;
}

union Frag { v16bf v; v8bf h[2]; };

#define WMMA_BF16(a, b, c) \
    __builtin_amdgcn_wmma_f32_16x16x32_bf16(false, (a), false, (b), (short)0, (c), false, false)

// async global->LDS staging (ASYNCcnt-tracked); GV mode: 64-bit vaddr
__device__ __forceinline__ void async_ld_b128(unsigned lds_off, const void* gaddr) {
    asm volatile("global_load_async_to_lds_b128 %0, %1, off"
                 :: "v"(lds_off), "v"((unsigned long long)gaddr) : "memory");
}
__device__ __forceinline__ void wait_async0() {
    asm volatile("s_wait_asynccnt 0" ::: "memory");
}

// ---------- prep: f32 -> bf16 ----------
__global__ __launch_bounds__(256) void cvt_bf(const float* __restrict__ in,
                                              __bf16* __restrict__ out, int n) {
    int i = blockIdx.x * 256 + threadIdx.x;
    if (i < n) out[i] = f2bf(in[i]);
}

// ---------- prep: W[k][n] -> WT[n][k] bf16 ----------
__global__ __launch_bounds__(256) void transpose_bf(const float* __restrict__ W,
                                                    __bf16* __restrict__ WT) {
    int n = blockIdx.x * 16 + threadIdx.x;
    int k = blockIdx.y * 16 + threadIdx.y;
    WT[(size_t)n * Ez + k] = f2bf(W[(size_t)k * Ez + n]);
}

// ---------- GEMM: C[4096,512] = A(bf16) * BT^T + bias, 16x64 per wave ----------
// mode 0: store bf16 [B,H,S,D]   (Q, K)
// mode 1: store bf16 [B,H,D,S]   (V transposed)
// mode 2: store f32  [M,N]       (final output)
__global__ __launch_bounds__(256)
void gemm16(const __bf16* __restrict__ A, const __bf16* __restrict__ BT,
            const float* __restrict__ bias, void* __restrict__ outp, int mode)
{
    const int K = Ez, N = Ez;
    int wave = threadIdx.x >> 5;
    int lane = threadIdx.x & 31;
    int lo = lane & 15, hi = lane >> 4;
    int tile = blockIdx.x * 8 + wave;    // 2048 wave-tiles = 256(M) x 8(N/64)
    int tm  = tile >> 3;
    int tns = tile & 7;                  // 64-wide N supertile

    const __bf16* Ap  = A  + (size_t)(tm * 16 + lo) * K + hi * 16;
    const __bf16* Bp0 = BT + (size_t)(tns * 64 + lo) * K + hi * 16;

    v8f c0 = {}, c1 = {}, c2 = {}, c3 = {};
    #pragma unroll 4
    for (int kc = 0; kc < K; kc += 32) {
        Frag a, b0, b1, b2, b3;
        a.h[0]  = *(const v8bf*)(Ap + kc);               // K = hi*16 + 0..7
        a.h[1]  = *(const v8bf*)(Ap + kc + 8);           // K = hi*16 + 8..15
        b0.h[0] = *(const v8bf*)(Bp0 + kc);              // K = hi*16 + 0..7
        b0.h[1] = *(const v8bf*)(Bp0 + kc + 8);          // K = hi*16 + 8..15
        b1.h[0] = *(const v8bf*)(Bp0 + (size_t)16 * K + kc);
        b1.h[1] = *(const v8bf*)(Bp0 + (size_t)16 * K + kc + 8);
        b2.h[0] = *(const v8bf*)(Bp0 + (size_t)32 * K + kc);
        b2.h[1] = *(const v8bf*)(Bp0 + (size_t)32 * K + kc + 8);
        b3.h[0] = *(const v8bf*)(Bp0 + (size_t)48 * K + kc);
        b3.h[1] = *(const v8bf*)(Bp0 + (size_t)48 * K + kc + 8);
        c0 = WMMA_BF16(a.v, b0.v, c0);
        c1 = WMMA_BF16(a.v, b1.v, c1);
        c2 = WMMA_BF16(a.v, b2.v, c2);
        c3 = WMMA_BF16(a.v, b3.v, c3);
    }

    #pragma unroll
    for (int nt = 0; nt < 4; nt++) {
        int bcol = tns * 64 + nt * 16 + lo;
        float bv = bias[bcol];
        v8f cc = (nt == 0) ? c0 : (nt == 1) ? c1 : (nt == 2) ? c2 : c3;
        if (mode == 2) {
            float* out = (float*)outp;
            #pragma unroll
            for (int r = 0; r < 8; r++) {
                int m = tm * 16 + r + hi * 8;
                out[(size_t)m * N + bcol] = cc[r] + bv;
            }
        } else {
            __bf16* out = (__bf16*)outp;
            int h = bcol >> 6, d = bcol & 63;
            #pragma unroll
            for (int r = 0; r < 8; r++) {
                int m = tm * 16 + r + hi * 8;
                int bb = m >> 11, s = m & 2047;
                size_t idx = (mode == 0)
                    ? ((size_t)((bb * Hz + h) * Sz + s)) * Dz + d
                    : ((size_t)((bb * Hz + h) * Dz + d)) * Sz + s;
                out[idx] = f2bf(cc[r] + bv);
            }
        }
    }
}

// ---------- flash attention ----------
// Block = 4 waves, all same (b,h); K/V 64-key tiles staged once per block into
// LDS with async loads, shared by all 4 waves (4x global traffic reduction).
#define AW 4
#define LPAD 72   // row stride in bf16: 144 B = 36 dwords -> conflict-free b128
__global__ __launch_bounds__(128)
void attn(const __bf16* __restrict__ Q, const __bf16* __restrict__ Kc,
          const __bf16* __restrict__ Vt, __bf16* __restrict__ Ctx)
{
    __shared__ __align__(16) __bf16 Klds[64][LPAD];   // 9 KB: keys x d
    __shared__ __align__(16) __bf16 Vlds[64][LPAD];   // 9 KB: d x keys
    __shared__ __align__(16) float  sc[AW][16][65];
    __shared__ __align__(16) __bf16 pb[AW][16][LPAD];
    __shared__ __align__(16) float  sred[AW][16];

    int w = threadIdx.x >> 5, lane = threadIdx.x & 31;
    int lo = lane & 15, hi = lane >> 4;
    int bh = blockIdx.x >> 5;                      // b*H + h
    int qt = (blockIdx.x & 31) * AW + w;           // query tile 0..127
    const float scale = 0.022097086912079608f;     // 1/sqrt(S) per reference

    // Q A-fragments (rows = 16 queries, K-dim = d, two frags cover d=0..63)
    const __bf16* Qp = Q + (((size_t)bh * Sz) + qt * 16 + lo) * Dz + hi * 16;
    Frag qa0, qa1;
    qa0.h[0] = *(const v8bf*)(Qp);
    qa0.h[1] = *(const v8bf*)(Qp + 8);
    qa1.h[0] = *(const v8bf*)(Qp + 32);
    qa1.h[1] = *(const v8bf*)(Qp + 40);

    v8f acc0 = {}, acc1 = {}, acc2 = {}, acc3 = {};
    float m_i = -3.0e38f, l_i = 0.f;

    const __bf16* Kp = Kc + ((size_t)bh * Sz) * Dz;
    const __bf16* Vp = Vt + ((size_t)bh * Dz) * Sz;
    unsigned kbase = (unsigned)(size_t)&Klds[0][0];
    unsigned vbase = (unsigned)(size_t)&Vlds[0][0];

    for (int kt = 0; kt < Sz; kt += 64) {
        __syncthreads();   // previous iteration's readers of Klds/Vlds are done
        // ---- cooperative async staging: 8 KB K-tile + 8 KB V-tile ----
        #pragma unroll
        for (int q = 0; q < 4; q++) {
            int c = q * 128 + threadIdx.x;         // 0..511 : 16B chunks
            int row = c >> 3, part = c & 7;
            async_ld_b128(kbase + row * (LPAD * 2) + part * 16,
                          Kp + (size_t)(kt + row) * Dz + part * 8);
        }
        #pragma unroll
        for (int q = 0; q < 4; q++) {
            int c = q * 128 + threadIdx.x;
            int row = c >> 3, part = c & 7;
            async_ld_b128(vbase + row * (LPAD * 2) + part * 16,
                          Vp + (size_t)row * Sz + kt + part * 8);
        }
        wait_async0();
        __syncthreads();   // tiles visible to all waves

        // ---- scores: 4 key subtiles of 16, K-dim = d (2 WMMAs each) ----
        #pragma unroll
        for (int j = 0; j < 4; j++) {
            const __bf16* kp = &Klds[j * 16 + lo][hi * 16];
            Frag b0, b1;
            b0.h[0] = *(const v8bf*)(kp);
            b0.h[1] = *(const v8bf*)(kp + 8);
            b1.h[0] = *(const v8bf*)(kp + 32);
            b1.h[1] = *(const v8bf*)(kp + 40);
            v8f s8 = {};
            s8 = WMMA_BF16(qa0.v, b0.v, s8);
            s8 = WMMA_BF16(qa1.v, b1.v, s8);
            #pragma unroll
            for (int r = 0; r < 8; r++)
                sc[w][r + hi * 8][j * 16 + lo] = s8[r] * scale;
        }

        // ---- online softmax: lane L (<16) owns query row L ----
        if (lane < 16) {
            int row = lane;
            float cmax = -3.0e38f;
            #pragma unroll 8
            for (int cc = 0; cc < 64; cc++) cmax = fmaxf(cmax, sc[w][row][cc]);
            float nm = fmaxf(m_i, cmax);
            float al = __expf(m_i - nm);
            float ps = 0.f;
            #pragma unroll 8
            for (int cc = 0; cc < 64; cc++) {
                float p = __expf(sc[w][row][cc] - nm);
                ps += p;
                pb[w][row][cc] = f2bf(p);
            }
            l_i = al * l_i + ps;
            m_i = nm;
            sred[w][row] = al;
        }
        // rescale running context (same-wave LDS is in-order)
        #pragma unroll
        for (int r = 0; r < 8; r++) {
            float al = sred[w][r + hi * 8];
            acc0[r] *= al; acc1[r] *= al; acc2[r] *= al; acc3[r] *= al;
        }

        // ---- P(16x64) x V(64x64): P A-frags + V B-frags from LDS ----
        Frag pa0, pa1;
        pa0.h[0] = *(const v8bf*)&pb[w][lo][hi * 16];
        pa0.h[1] = *(const v8bf*)&pb[w][lo][hi * 16 + 8];
        pa1.h[0] = *(const v8bf*)&pb[w][lo][hi * 16 + 32];
        pa1.h[1] = *(const v8bf*)&pb[w][lo][hi * 16 + 40];
        #pragma unroll
        for (int dt = 0; dt < 4; dt++) {
            const __bf16* vp = &Vlds[dt * 16 + lo][hi * 16];
            Frag vb0, vb1;
            vb0.h[0] = *(const v8bf*)(vp);
            vb0.h[1] = *(const v8bf*)(vp + 8);
            vb1.h[0] = *(const v8bf*)(vp + 32);
            vb1.h[1] = *(const v8bf*)(vp + 40);
            v8f a = (dt == 0) ? acc0 : (dt == 1) ? acc1 : (dt == 2) ? acc2 : acc3;
            a = WMMA_BF16(pa0.v, vb0.v, a);
            a = WMMA_BF16(pa1.v, vb1.v, a);
            if (dt == 0) acc0 = a; else if (dt == 1) acc1 = a;
            else if (dt == 2) acc2 = a; else acc3 = a;
        }
    }

    // ---- finalize: divide by row sums, store ctx bf16 [B,S,(h,d)] ----
    if (lane < 16) sred[w][lane] = (l_i > 0.f) ? (1.f / l_i) : 0.f;
    int b = bh >> 3, h = bh & 7;
    #pragma unroll
    for (int r = 0; r < 8; r++) {
        float inv = sred[w][r + hi * 8];
        size_t orow = ((size_t)b * Sz + qt * 16 + r + hi * 8) * Ez + h * Dz;
        Ctx[orow +  0 + lo] = f2bf(acc0[r] * inv);
        Ctx[orow + 16 + lo] = f2bf(acc1[r] * inv);
        Ctx[orow + 32 + lo] = f2bf(acc2[r] * inv);
        Ctx[orow + 48 + lo] = f2bf(acc3[r] * inv);
    }
}

extern "C" void kernel_launch(void* const* d_in, const int* in_sizes, int n_in,
                              void* d_out, int out_size, void* d_ws, size_t ws_size,
                              hipStream_t stream) {
    (void)in_sizes; (void)n_in; (void)out_size; (void)ws_size;
    const float* x  = (const float*)d_in[0];
    const float* Wq = (const float*)d_in[1];
    const float* bq = (const float*)d_in[2];
    const float* Wk = (const float*)d_in[3];
    const float* bk = (const float*)d_in[4];
    const float* Wv = (const float*)d_in[5];
    const float* bv = (const float*)d_in[6];
    const float* Wo = (const float*)d_in[7];
    const float* bo = (const float*)d_in[8];

    char* ws = (char*)d_ws;
    const size_t MB = 1u << 20;
    __bf16* xb  = (__bf16*)(ws);                       // 4 MB [4096,512]
    __bf16* Qb  = (__bf16*)(ws +  4 * MB);             // 4 MB [B,H,S,D]
    __bf16* Kb  = (__bf16*)(ws +  8 * MB);             // 4 MB [B,H,S,D]
    __bf16* Vt  = (__bf16*)(ws + 12 * MB);             // 4 MB [B,H,D,S]
    __bf16* Ctx = (__bf16*)(ws + 16 * MB);             // 4 MB [4096,512]
    __bf16* WqT = (__bf16*)(ws + 20 * MB);
    __bf16* WkT = (__bf16*)(ws + 20 * MB + 512 * 1024);
    __bf16* WvT = (__bf16*)(ws + 21 * MB);
    __bf16* WoT = (__bf16*)(ws + 21 * MB + 512 * 1024);

    cvt_bf<<<ROWS * Ez / 256, 256, 0, stream>>>(x, xb, ROWS * Ez);
    dim3 tb(16, 16), tg(Ez / 16, Ez / 16);
    transpose_bf<<<tg, tb, 0, stream>>>(Wq, WqT);
    transpose_bf<<<tg, tb, 0, stream>>>(Wk, WkT);
    transpose_bf<<<tg, tb, 0, stream>>>(Wv, WvT);
    transpose_bf<<<tg, tb, 0, stream>>>(Wo, WoT);

    gemm16<<<256, 256, 0, stream>>>(xb, WqT, bq, Qb, 0);
    gemm16<<<256, 256, 0, stream>>>(xb, WkT, bk, Kb, 0);
    gemm16<<<256, 256, 0, stream>>>(xb, WvT, bv, Vt, 1);

    attn<<<(Bz * Hz * (Sz / 16)) / AW, AW * 32, 0, stream>>>(Qb, Kb, Vt, Ctx);

    gemm16<<<256, 256, 0, stream>>>(Ctx, WoT, bo, d_out, 2);
}
